// contrastive_Loss_76991583748731
// MI455X (gfx1250) — matrix operand order repaired and checked by reading, not compiled, not run on hardware
//
#include <hip/hip_runtime.h>

// ---------------- problem constants (match reference) ----------------
#define NS    1048576            // samples
#define DF    128                // feature dim
#define NC    100                // classes
#define CPAD  112                // classes padded to 7 tiles of 16
#define DPAD  136                // anchor LDS row stride (ushorts): 272B => b128 loads bank-spread
#define SPAD  113                // score LDS row stride (floats): odd => conflict-free
#define NWAVE 8
#define TPB   (NWAVE * 32)       // 256 threads, 8 wave32
#define SAMP_PER_BLOCK (NWAVE * 16)   // 128 samples / block
#define NBLK  (NS / SAMP_PER_BLOCK)   // 8192 blocks

// LDS layout (bytes)
#define OFF_BHI   0
#define OFF_BLO   (CPAD * DPAD * 2)                    // 30464
#define OFF_SCORE (OFF_BLO * 2)                        // 60928
#define OFF_CE    (OFF_SCORE + NWAVE * 16 * SPAD * 4)  // 60928 + 57856 = 118784
#define SMEM_TOTAL (OFF_CE + NWAVE * 16 * 4)           // 119296

// workspace layout (bytes)
#define WS_PRESENT_OFF 0                 // 112 ints
#define WS_BLKSUM_OFF  512               // NBLK floats

typedef __attribute__((ext_vector_type(16))) __bf16 v16bf;
typedef __attribute__((ext_vector_type(8)))  float  v8f;

union BF16Frag {
  v16bf v;
  __bf16 b[16];
  uint4 q[2];
};

// x = hi + lo with hi = bf16_rne(x), lo = bf16_rne(x - hi): ~f32 accuracy when
// recombined as hi*hi + hi*lo + lo*hi in WMMA (dropped lo*lo term ~2^-18 rel).
// Native casts let clang emit v_cvt_pk_bf16_f32 instead of manual RNE bit-ops.
__device__ __forceinline__ void bf16_split(float f, __bf16& hi, __bf16& lo) {
  hi = (__bf16)f;
  lo = (__bf16)(f - (float)hi);
}

// --------------------------------------------------------------------
__global__ void ce_init_kernel(int* present) {
  if (threadIdx.x < CPAD) present[threadIdx.x] = 0;
}

// --------------------------------------------------------------------
__global__ __launch_bounds__(TPB)
void ce_main_kernel(const float* __restrict__ x,
                    const float* __restrict__ anchors,
                    const int*   __restrict__ y,
                    float*       __restrict__ blockSums,
                    int*         __restrict__ present) {
  extern __shared__ unsigned char smem[];
  __bf16* sBhi   = (__bf16*)(smem + OFF_BHI);
  __bf16* sBlo   = (__bf16*)(smem + OFF_BLO);
  float*  sScore = (float*)(smem + OFF_SCORE);
  float*  sCe    = (float*)(smem + OFF_CE);

  const int tid = threadIdx.x;

  // ---- stage anchors into LDS as split bf16 (hi/lo), zero-pad classes 100..111.
  // Anchors are 51 KB and resident in the 192 MB L2, so the per-block re-read is free.
  for (int i = tid; i < CPAD * DF; i += TPB) {
    int c = i >> 7;          // /128
    int k = i & (DF - 1);
    float v = (c < NC) ? anchors[c * DF + k] : 0.0f;
    __bf16 hi, lo;
    bf16_split(v, hi, lo);
    sBhi[c * DPAD + k] = hi;
    sBlo[c * DPAD + k] = lo;
  }
  __syncthreads();

  const int wave = tid >> 5;
  const int lane = tid & 31;
  const int h    = lane >> 4;   // half-wave selects K sub-ranges (A) / K half (B)
  const int m    = lane & 15;   // A row (sample) / B column (class) within tile
  const int srow = wave * 16 + m;                        // sample row this lane owns
  const size_t row = (size_t)blockIdx.x * SAMP_PER_BLOCK + srow;
  const float* xr = x + row * DF;

  v8f acc[7];
#pragma unroll
  for (int t = 0; t < 7; ++t) {
#pragma unroll
    for (int e = 0; e < 8; ++e) acc[t][e] = 0.0f;
  }

  // ---- GEMM: scores[16 x 112] = x_tile[16 x 128] * anchors^T, bf16x3 split precision
#pragma unroll
  for (int k = 0; k < 4; ++k) {
    const int kb = 32 * k;
    // A fragment: lane(h,m) holds K = [16h,16h+16) of row m (matches B)
    const float4* p0 = (const float4*)(xr + kb + 16 * h);
    const float4* p1 = (const float4*)(xr + kb + 16 * h + 8);
    float4 a0 = p0[0], a1 = p0[1];
    float4 b0 = p1[0], b1 = p1[1];
    float f[16] = {a0.x, a0.y, a0.z, a0.w, a1.x, a1.y, a1.z, a1.w,
                   b0.x, b0.y, b0.z, b0.w, b1.x, b1.y, b1.z, b1.w};
    BF16Frag Ahi, Alo;
#pragma unroll
    for (int e = 0; e < 16; ++e) bf16_split(f[e], Ahi.b[e], Alo.b[e]);

#pragma unroll
    for (int t = 0; t < 7; ++t) {
      // B fragment: lane(h,m) holds K = [kb+16h, kb+16h+16) of anchor column 16t+m
      const int cls = t * 16 + m;
      const uint4* bh = (const uint4*)&sBhi[cls * DPAD + kb + 16 * h];
      const uint4* bl = (const uint4*)&sBlo[cls * DPAD + kb + 16 * h];
      BF16Frag Bhi, Blo;
      Bhi.q[0] = bh[0]; Bhi.q[1] = bh[1];
      Blo.q[0] = bl[0]; Blo.q[1] = bl[1];
      acc[t] = __builtin_amdgcn_wmma_f32_16x16x32_bf16(false, Ahi.v, false, Bhi.v,
                                                       (short)0, acc[t], false, false);
      acc[t] = __builtin_amdgcn_wmma_f32_16x16x32_bf16(false, Ahi.v, false, Blo.v,
                                                       (short)0, acc[t], false, false);
      acc[t] = __builtin_amdgcn_wmma_f32_16x16x32_bf16(false, Alo.v, false, Bhi.v,
                                                       (short)0, acc[t], false, false);
    }
  }

  // ---- spill scores to LDS. C/D layout: VGPR r => sample r+8h, lane m => class 16t+m.
  // Padded classes 100..111 land in LDS but are never read below.
#pragma unroll
  for (int t = 0; t < 7; ++t) {
#pragma unroll
    for (int r = 0; r < 8; ++r) {
      sScore[(wave * 16 + r + 8 * h) * SPAD + t * 16 + m] = acc[t][r];
    }
  }
  __syncthreads();

  // ---- log-softmax + CE. Lane(h,m) handles sample m's classes [50h, 50h+50),
  // halves combined across the wave32 with shfl_xor(16).
  const float* sc = &sScore[srow * SPAD];
  const int c0 = h * 50, c1 = c0 + 50;
  float mx = -3.0e38f;
  for (int c = c0; c < c1; ++c) mx = fmaxf(mx, sc[c]);
  float M = fmaxf(mx, __shfl_xor(mx, 16));
  float sm = 0.0f;
  for (int c = c0; c < c1; ++c) sm += __expf(sc[c] - M);
  float S = sm + __shfl_xor(sm, 16);

  const int label = y[row];
  const float ce = (M + __logf(S)) - sc[label];

  if (h == 0) {
    sCe[srow] = ce;
    present[label] = 1;   // idempotent store; races benign
  }
  __syncthreads();

  // fixed-order per-block reduction => deterministic output
  if (tid == 0) {
    float tsum = 0.0f;
    for (int i = 0; i < SAMP_PER_BLOCK; ++i) tsum += sCe[i];
    blockSums[blockIdx.x] = tsum;
  }
}

// --------------------------------------------------------------------
__global__ void ce_finalize_kernel(const float* __restrict__ blockSums,
                                   const int*   __restrict__ present,
                                   float*       __restrict__ out) {
  __shared__ float part[256];
  const int tid = threadIdx.x;
  float s = 0.0f;
  for (int i = tid; i < NBLK; i += 256) s += blockSums[i];   // fixed per-thread order
  part[tid] = s;
  __syncthreads();
  if (tid == 0) {
    float tot = 0.0f;
    for (int i = 0; i < 256; ++i) tot += part[i];            // fixed order
    int cnt = 0;
    for (int c = 0; c < NC; ++c) cnt += (present[c] != 0);
    out[0] = (tot / (float)NS) * (float)cnt;
  }
}

// --------------------------------------------------------------------
extern "C" void kernel_launch(void* const* d_in, const int* in_sizes, int n_in,
                              void* d_out, int out_size, void* d_ws, size_t ws_size,
                              hipStream_t stream) {
  const float* x       = (const float*)d_in[0];
  const float* anchors = (const float*)d_in[1];
  const int*   y       = (const int*)d_in[2];
  float*       out     = (float*)d_out;

  int*   present   = (int*)((char*)d_ws + WS_PRESENT_OFF);
  float* blockSums = (float*)((char*)d_ws + WS_BLKSUM_OFF);

  (void)in_sizes; (void)n_in; (void)out_size; (void)ws_size;

  hipFuncSetAttribute((const void*)ce_main_kernel,
                      hipFuncAttributeMaxDynamicSharedMemorySize, SMEM_TOTAL);

  ce_init_kernel<<<1, 128, 0, stream>>>(present);
  ce_main_kernel<<<NBLK, TPB, SMEM_TOTAL, stream>>>(x, anchors, y, blockSums, present);
  ce_finalize_kernel<<<1, 256, 0, stream>>>(blockSums, present, out);
}
